// UltraEfficientSparseFFN_19069654794531
// MI455X (gfx1250) — matrix operand order repaired and checked by reading, not compiled, not run on hardware
//
#include <hip/hip_runtime.h>

#define HIDDEN 1024
#define FFN    4096
#define TOPK   32
#define TPB    16            // tokens per block
#define XROWB  2064          // padded x-tile row stride in bytes (2048 + 8B TDM pad per 1024B)

typedef __bf16 v4bf  __attribute__((ext_vector_type(4)));
typedef __bf16 v8bf  __attribute__((ext_vector_type(8)));
typedef __bf16 v16bf __attribute__((ext_vector_type(16)));
typedef float  v8f   __attribute__((ext_vector_type(8)));
typedef unsigned int v4u __attribute__((ext_vector_type(4)));
typedef int    v4i   __attribute__((ext_vector_type(4)));
typedef int    v8i   __attribute__((ext_vector_type(8)));

union ABFrag { v16bf v; v8bf h[2]; v4bf q[4]; };

// ---------------- prep: f32 -> bf16 (used for W1 and x) ----------------
__global__ __launch_bounds__(256) void cvt_f32_bf16(const float* __restrict__ in,
                                                    __bf16* __restrict__ outp) {
  int i = (blockIdx.x * 256 + threadIdx.x) * 4;
  float4 v = *(const float4*)(in + i);
  v4bf b;
  b[0] = (__bf16)v.x; b[1] = (__bf16)v.y; b[2] = (__bf16)v.z; b[3] = (__bf16)v.w;
  *(v4bf*)(outp + i) = b;
}

// ---------------- prep: W2 [HIDDEN][FFN] -> W2T [FFN][HIDDEN] ----------------
__global__ __launch_bounds__(256) void transpose_w2(const float* __restrict__ W2,
                                                    float* __restrict__ W2T) {
  __shared__ float tile[32][33];
  const int bx = blockIdx.x & 127;   // FFN tile   (FFN/32 = 128)
  const int by = blockIdx.x >> 7;    // HIDDEN tile (HIDDEN/32 = 32)
  const int tx = threadIdx.x & 31;
  const int ty = threadIdx.x >> 5;   // 0..7
  const int f0 = bx * 32, h0 = by * 32;
#pragma unroll
  for (int i = 0; i < 32; i += 8)
    tile[ty + i][tx] = W2[(size_t)(h0 + ty + i) * FFN + f0 + tx];
  __syncthreads();
#pragma unroll
  for (int i = 0; i < 32; i += 8)
    W2T[(size_t)(f0 + ty + i) * HIDDEN + h0 + tx] = tile[tx][ty + i];
}

// ---------------- fused FFN: TDM stage + WMMA GEMM + top-32 + cubic + sparse GEMV ----------------
__global__ __launch_bounds__(256, 1) void ffn_fused(
    const __bf16* __restrict__ xb, const float* __restrict__ b1,
    const float* __restrict__ b2, const __bf16* __restrict__ W1b,
    const float* __restrict__ W2T, float* __restrict__ out) {
  extern __shared__ char smem[];
  // xs: 16 rows x 2064 B (bf16 row + 8B pad after each 1024B half) = 33024 B
  float* hs = (float*)(smem + TPB * XROWB);                       // 16 x 4096 f32
  float* sv = (float*)(smem + TPB * XROWB + TPB * FFN * 4);       // 16 x 32
  int*   si = (int*)  (smem + TPB * XROWB + TPB * FFN * 4 + TPB * TOPK * 4);

  const int tid  = threadIdx.x;
  const int lane = tid & 31;
  const int wave = tid >> 5;
  const int t0   = blockIdx.x * TPB;

  // ---- Phase A: stage 16x1024 bf16 x tile into LDS via Tensor Data Mover ----
#if __has_builtin(__builtin_amdgcn_tensor_load_to_lds) && __has_builtin(__builtin_amdgcn_s_wait_tensorcnt)
  if (wave == 0) {
    const unsigned lds_addr = (unsigned)(size_t)smem;   // low 32 bits = LDS byte offset
    const unsigned long long ga = (unsigned long long)(const void*)(xb + (size_t)t0 * HIDDEN);
    // D# group0: count=1 | lds_addr | global_addr[56:0] | type=2
    v4u g0 = { 1u, lds_addr, (unsigned)(ga & 0xffffffffu),
               (unsigned)((ga >> 32) & 0x01ffffffu) | (2u << 30) };
    // D# group1: data_size=3 (8B), pad_enable, pad_interval=7 (256 dw), pad_amount=1 (2 dw)
    // tensor_dim0 = tile_dim0 = stride0 = 256 (8B units = 2048 B row), 16 rows
    v8i g1 = { (int)((3u << 16) | (1u << 20) | (7u << 22) | (1u << 25)),
               (int)(256u << 16),          // tensor_dim0[15:0] << 16
               (int)(16u << 16),           // tensor_dim1[15:0] << 16
               (int)(256u << 16),          // tile_dim0 << 16
               16,                         // tile_dim1 = 16, tile_dim2 = 0
               256,                        // tensor_dim0_stride[31:0]
               0, 0 };
    v4i gz = { 0, 0, 0, 0 };
#if defined(__clang_major__) && (__clang_major__ >= 23)
    v8i gz8 = { 0, 0, 0, 0, 0, 0, 0, 0 };
    __builtin_amdgcn_tensor_load_to_lds(g0, g1, gz, gz, gz8, 0);
#else
    __builtin_amdgcn_tensor_load_to_lds(g0, g1, gz, gz, 0);
#endif
    __builtin_amdgcn_s_wait_tensorcnt(0);
  }
#else
  // Fallback: manual staging with identical padded layout
  {
    const __bf16* xsrc = xb + (size_t)t0 * HIDDEN;
    for (int g = tid; g < TPB * HIDDEN / 8; g += 256) {
      const int row = g >> 7;              // 128 8-elem groups per row
      const int col = (g & 127) * 8;
      v8bf v = *(const v8bf*)(xsrc + row * HIDDEN + col);
      char* dst = smem + row * XROWB + col * 2 + ((col >= 512) ? 8 : 0);
      *(v4bf*)dst       = *(v4bf*)&v;
      *(v4bf*)(dst + 8) = *((v4bf*)&v + 1);
    }
  }
#endif
  __syncthreads();

  // ---- Phase B: h = x @ W1^T + b1 via v_wmma_f32_16x16x32_bf16, 2 N-tiles/wave ----
  const int mrow  = lane & 15;
  const int khalf = lane >> 4;
  const char* arowb = smem + mrow * XROWB;
  for (int chunk = 0; chunk < FFN / 256; ++chunk) {   // 16 chunks of 256 N
    const int n0 = chunk * 256 + wave * 32 + (lane & 15);
    const int n1 = n0 + 16;
    const __bf16* brow0 = W1b + (size_t)n0 * HIDDEN;
    const __bf16* brow1 = W1b + (size_t)n1 * HIDDEN;
    v8f c0 = {}, c1 = {};
#pragma unroll
    for (int k0 = 0; k0 < HIDDEN; k0 += 32) {
      const int pad = (k0 >= 512) ? 8 : 0;            // compile-time per unrolled step
      const char* ap = arowb + 2 * k0 + 32 * khalf + pad;
      ABFrag a, bf0, bf1;
      // A 16x32, contiguous k order chosen here (not ISA table order): lanes<16 hold K{0..15}, lanes>=16 hold K{16..31} (matches B)
      a.q[0] = *(const v4bf*)(ap);
      a.q[1] = *(const v4bf*)(ap + 8);
      a.q[2] = *(const v4bf*)(ap + 16);
      a.q[3] = *(const v4bf*)(ap + 24);
      // B 32x16: column n per lane; half-wave K split, contiguous in W1 row
      bf0.v = *(const v16bf*)(brow0 + k0 + khalf * 16);
      bf1.v = *(const v16bf*)(brow1 + k0 + khalf * 16);
      c0 = __builtin_amdgcn_wmma_f32_16x16x32_bf16(false, a.v, false, bf0.v,
                                                   (short)0, c0, false, false);
      c1 = __builtin_amdgcn_wmma_f32_16x16x32_bf16(false, a.v, false, bf1.v,
                                                   (short)0, c1, false, false);
    }
    const float bias0 = b1[n0];
    const float bias1 = b1[n1];
#pragma unroll
    for (int r = 0; r < 8; ++r) {
      const int m = r + khalf * 8;       // C/D layout: VGPR r, lane half selects M / M+8
      hs[m * FFN + n0] = c0[r] + bias0;
      hs[m * FFN + n1] = c1[r] + bias1;
    }
  }
  __syncthreads();

  // ---- Phase C: per-token top-32 by |h|, record h^3 and index, zero winner ----
  for (int tt = 0; tt < 2; ++tt) {
    const int t = wave * 2 + tt;         // 8 waves x 2 tokens = 16
    float* hrow = hs + t * FFN;
    for (int s = 0; s < TOPK; ++s) {
      float besta = -1.0f, bestv = 0.0f;
      int   besti = 0;
#pragma unroll
      for (int i = 0; i < FFN / 128; ++i) {
        const int j = i * 128 + lane * 4;
        float4 v = *(const float4*)(hrow + j);
        float a0 = fabsf(v.x); if (a0 > besta) { besta = a0; bestv = v.x; besti = j; }
        float a1 = fabsf(v.y); if (a1 > besta) { besta = a1; bestv = v.y; besti = j + 1; }
        float a2 = fabsf(v.z); if (a2 > besta) { besta = a2; bestv = v.z; besti = j + 2; }
        float a3 = fabsf(v.w); if (a3 > besta) { besta = a3; bestv = v.w; besti = j + 3; }
      }
#pragma unroll
      for (int off = 16; off > 0; off >>= 1) {
        float oa = __shfl_xor(besta, off);
        float ov = __shfl_xor(bestv, off);
        int   oi = __shfl_xor(besti, off);
        if (oa > besta || (oa == besta && oi < besti)) { besta = oa; bestv = ov; besti = oi; }
      }
      if (lane == 0) {
        sv[t * TOPK + s] = bestv * bestv * bestv;  // cubic activation
        si[t * TOPK + s] = besti;
        hrow[besti] = 0.0f;                        // remove winner for next round
      }
    }
  }
  __syncthreads();

  // ---- Phase D: out = sum_j h^3_j * W2T[j,:] + b2 (coalesced) ----
  const int hid = tid * 4;               // 256 threads x float4 = 1024
  const float4 bb = *(const float4*)(b2 + hid);
  for (int t = 0; t < TPB; ++t) {
    float4 acc = bb;
#pragma unroll 8
    for (int s = 0; s < TOPK; ++s) {
      const float v = sv[t * TOPK + s];
      const int   j = si[t * TOPK + s];
      const float4 w = *(const float4*)(W2T + (size_t)j * HIDDEN + hid);
      acc.x += v * w.x; acc.y += v * w.y; acc.z += v * w.z; acc.w += v * w.w;
    }
    *(float4*)(out + (size_t)(t0 + t) * HIDDEN + hid) = acc;
  }
}

extern "C" void kernel_launch(void* const* d_in, const int* in_sizes, int n_in,
                              void* d_out, int out_size, void* d_ws, size_t ws_size,
                              hipStream_t stream) {
  (void)in_sizes; (void)n_in; (void)out_size; (void)ws_size;
  const float* x  = (const float*)d_in[0];
  const float* W1 = (const float*)d_in[1];
  const float* b1 = (const float*)d_in[2];
  const float* W2 = (const float*)d_in[3];
  const float* b2 = (const float*)d_in[4];
  float* out = (float*)d_out;

  const int tokens = 4 * 2048;
  __bf16* W1b = (__bf16*)d_ws;                                            // 8 MB
  float*  W2T = (float*)((char*)d_ws + (size_t)FFN * HIDDEN * 2);         // 16 MB
  __bf16* xbf = (__bf16*)((char*)d_ws + (size_t)FFN * HIDDEN * 2
                                      + (size_t)FFN * HIDDEN * 4);        // 16 MB

  cvt_f32_bf16<<<FFN * HIDDEN / 1024, 256, 0, stream>>>(W1, W1b);
  cvt_f32_bf16<<<tokens * HIDDEN / 1024, 256, 0, stream>>>(x, xbf);
  transpose_w2<<<(HIDDEN / 32) * (FFN / 32), 256, 0, stream>>>(W2, W2T);

  const size_t shmem = (size_t)TPB * XROWB      // padded x tile (bf16)
                     + (size_t)TPB * FFN * 4    // h buffer (f32)
                     + (size_t)TPB * TOPK * 8;  // selected values + indices
  ffn_fused<<<tokens / TPB, 256, shmem, stream>>>(xbf, b1, b2, W1b, W2T, out);
}
